// TemporalGCN_6966436954288
// MI455X (gfx1250) — matrix operand order repaired and checked by reading, not compiled, not run on hardware
//
#include <hip/hip_runtime.h>
#include <cstddef>

// ---------------- WMMA types ----------------
typedef __attribute__((ext_vector_type(16))) __bf16 v16bf;
typedef __attribute__((ext_vector_type(8)))  float  v8f;

__device__ __forceinline__ unsigned short f2bf(float f) {
    unsigned int u = __float_as_uint(f);
    u += 0x7FFFu + ((u >> 16) & 1u);      // round-to-nearest-even
    return (unsigned short)(u >> 16);
}

// ---------------- WMMA GEMM: C[M,N] = act(A[M,K] (bf16) * B[K,N] (bf16) + bias) ----------
// Block tile 128x64, 8 waves (wave32), wave tile 32x32 = 2x2 WMMA 16x16x32 tiles.
// A and B are pre-converted bf16; C is f32 or bf16 (c_bf16 flag). No conversion ALU in loop.
#define TBM 128
#define TBN 64
#define TBK 32
#define KPAD 40   // ushort row stride in LDS (padding vs bank conflicts)

__global__ __launch_bounds__(256)
void gemm_bf16_wmma(const unsigned short* __restrict__ A, const unsigned short* __restrict__ B,
                    const float* __restrict__ bias, void* __restrict__ Cout,
                    int M, int N, int K, int relu, int c_bf16)
{
    __shared__ unsigned short As[TBM * KPAD];
    __shared__ unsigned short Bs[TBN * KPAD];

    const int tid  = threadIdx.x;
    const int wave = tid >> 5;
    const int lane = tid & 31;
    const int wm   = wave & 3;      // 0..3 over M
    const int wn   = wave >> 2;     // 0..1 over N
    const int lrow = lane & 15;
    const int half = lane >> 4;     // 0/1

    const int m0 = blockIdx.x * TBM;
    const int n0 = blockIdx.y * TBN;

    // staging index precompute: A tile = 128x32 bf16 = 512 uint4, 2 per thread
    const int ar0 = tid >> 2;               // rows 0..63
    const int ag0 = (tid & 3) * 8;          // k-offset within tile
    const int ar1 = (tid + 256) >> 2;       // rows 64..127
    const int ag1 = ag0;
    // B tile = 32x64 bf16: thread -> (k, 8 consecutive n)
    const int bk  = tid >> 3;               // 0..31
    const int bn8 = (tid & 7) * 8;          // 0..56

    v8f acc[2][2] = {};

    for (int kt = 0; kt < K; kt += TBK) {
        // ---- issue ALL global loads for this tile first (batched loadcnt wait) ----
        int gr0 = m0 + ar0; if (gr0 >= M) gr0 = M - 1;
        int gr1 = m0 + ar1; if (gr1 >= M) gr1 = M - 1;
        const uint4 a0 = *(const uint4*)(A + (size_t)gr0 * K + kt + ag0);
        const uint4 a1 = *(const uint4*)(A + (size_t)gr1 * K + kt + ag1);
        union { uint4 q; unsigned short s[8]; } bu;
        bu.q = *(const uint4*)(B + (size_t)(kt + bk) * N + n0 + bn8);

        // prefetch next A tile (emits global_prefetch_b8)
        if (kt + TBK < K) {
            __builtin_prefetch(A + (size_t)gr0 * K + kt + TBK + ag0, 0, 1);
        }

        // ---- LDS stores ----
        {
            unsigned int* d0 = (unsigned int*)&As[ar0 * KPAD + ag0];
            d0[0] = a0.x; d0[1] = a0.y; d0[2] = a0.z; d0[3] = a0.w;
            unsigned int* d1 = (unsigned int*)&As[ar1 * KPAD + ag1];
            d1[0] = a1.x; d1[1] = a1.y; d1[2] = a1.z; d1[3] = a1.w;
            #pragma unroll
            for (int j = 0; j < 8; ++j) Bs[(bn8 + j) * KPAD + bk] = bu.s[j];   // transpose to [n][k]
        }
        __syncthreads();

        // ---- fragments (ISA 7.12.2 layouts) ----
        v16bf afrag[2], bfrag[2];
        #pragma unroll
        for (int tm = 0; tm < 2; ++tm) {
            int m = wm * 32 + tm * 16 + lrow;
            union { v16bf v; unsigned int u[8]; } fa;
            #pragma unroll
            for (int v8 = 0; v8 < 8; ++v8) {
                int pair = (v8 < 4 ? v8 : 8 + (v8 - 4)) + half * 4;   // K pairs: 0-7/8-15 lo-half, 16-23/24-31 hi
                fa.u[v8] = *(const unsigned int*)&As[m * KPAD + pair * 2];
            }
            afrag[tm] = fa.v;
        }
        #pragma unroll
        for (int tn = 0; tn < 2; ++tn) {
            int n = wn * 32 + tn * 16 + lrow;
            union { v16bf v; unsigned int u[8]; } fb;
            #pragma unroll
            for (int v8 = 0; v8 < 8; ++v8) {
                int k = (v8 < 4 ? 2 * v8 : 16 + 2 * (v8 - 4)) + half * 8;   // lanes 0-15: K 0-7,16-23; lanes 16-31: K 8-15,24-31
                fb.u[v8] = *(const unsigned int*)&Bs[n * KPAD + k];
            }
            bfrag[tn] = fb.v;
        }
        #pragma unroll
        for (int tm = 0; tm < 2; ++tm)
            #pragma unroll
            for (int tn = 0; tn < 2; ++tn)
                acc[tm][tn] = __builtin_amdgcn_wmma_f32_16x16x32_bf16(
                    false, afrag[tm], false, bfrag[tn], (short)0, acc[tm][tn], false, false);
        __syncthreads();
    }

    // ---- epilogue: bias + optional ReLU, f32 or bf16 store, guarded on M ----
    #pragma unroll
    for (int tm = 0; tm < 2; ++tm) {
        #pragma unroll
        for (int tn = 0; tn < 2; ++tn) {
            int n = n0 + wn * 32 + tn * 16 + lrow;
            float bv = bias ? bias[n] : 0.0f;
            #pragma unroll
            for (int r = 0; r < 8; ++r) {
                int m = m0 + wm * 32 + tm * 16 + half * 8 + r;   // C/D layout: VGPR r -> M (+8 hi half)
                if (m < M) {
                    float v = acc[tm][tn][r] + bv;
                    if (relu) v = fmaxf(v, 0.0f);
                    if (c_bf16) ((unsigned short*)Cout)[(size_t)m * N + n] = f2bf(v);
                    else        ((float*)Cout)[(size_t)m * N + n] = v;
                }
            }
        }
    }
}

// ---------------- utility / elementwise kernels ----------------
__global__ void k_init_f32(float* p, float val, int n) {
    int i = blockIdx.x * blockDim.x + threadIdx.x;
    if (i < n) p[i] = val;
}

__global__ void k_init_u16(unsigned short* p, unsigned short val, int n) {
    int i = blockIdx.x * blockDim.x + threadIdx.x;
    if (i < n) p[i] = val;
}

__global__ void k_conv_bf16(const float* __restrict__ src, unsigned short* __restrict__ dst, int n) {
    int i = blockIdx.x * blockDim.x + threadIdx.x;
    if (i < n) dst[i] = f2bf(src[i]);
}

// src [rows][cols] f32 -> dst [cols][rows] bf16 (for  x @ W.T  as  x @ (W^T) )
__global__ void k_conv_bf16_T(const float* __restrict__ src, unsigned short* __restrict__ dst,
                              int rows, int cols) {
    int i = blockIdx.x * blockDim.x + threadIdx.x;
    if (i >= rows * cols) return;
    int r = i / cols, c = i % cols;
    dst[(size_t)c * rows + r] = f2bf(src[i]);
}

__global__ void k_deg_accum(const int* __restrict__ dst, const float* __restrict__ w,
                            float* __restrict__ deg, int E) {
    int e = blockIdx.x * blockDim.x + threadIdx.x;
    if (e < E) atomicAdd(&deg[dst[e]], w[e]);
}

__global__ void k_dinv_self(const float* __restrict__ deg, float* __restrict__ dinv,
                            float* __restrict__ selfnorm, int n) {
    int i = blockIdx.x * blockDim.x + threadIdx.x;
    if (i >= n) return;
    float d = deg[i];
    float r = d > 0.0f ? rsqrtf(d) : 0.0f;
    dinv[i] = r;
    selfnorm[i] = r * r;
}

__global__ void k_edge_norm(const int* __restrict__ src, const int* __restrict__ dst,
                            const float* __restrict__ w, const float* __restrict__ dinv,
                            float* __restrict__ norm, int E) {
    int e = blockIdx.x * blockDim.x + threadIdx.x;
    if (e < E) norm[e] = dinv[src[e]] * w[e] * dinv[dst[e]];
}

// agg[i,:] = xw[i,:] * selfnorm[i]   (self-loop term, also initializes agg)
__global__ void k_scatter_init(const float* __restrict__ xw, const float* __restrict__ selfnorm,
                               float* __restrict__ agg, int n) {
    int idx = blockIdx.x * blockDim.x + threadIdx.x;     // over n*32 float4
    if (idx >= n * 32) return;
    int i = idx >> 5;
    float s = selfnorm[i];
    float4 v = ((const float4*)xw)[idx];
    v.x *= s; v.y *= s; v.z *= s; v.w *= s;
    ((float4*)agg)[idx] = v;
}

// agg[dst[e],:] += xw[src[e],:] * norm[e]
__global__ void k_scatter_edges(const int* __restrict__ src, const int* __restrict__ dst,
                                const float* __restrict__ norm, const float* __restrict__ xw,
                                float* __restrict__ agg, int E) {
    int idx = blockIdx.x * blockDim.x + threadIdx.x;     // over E*32 (float4 chunks)
    if (idx >= E * 32) return;
    int e = idx >> 5;
    int q = idx & 31;
    float nm = norm[e];                                   // uniform per wave
    const float4 v = ((const float4*)(xw + (size_t)src[e] * 128))[q];
    float* d = agg + (size_t)dst[e] * 128 + q * 4;
    atomicAdd(d + 0, v.x * nm);
    atomicAdd(d + 1, v.y * nm);
    atomicAdd(d + 2, v.z * nm);
    atomicAdd(d + 3, v.w * nm);
}

// per-channel sum & sumsq over rows (H=128); blockDim=128, grid-stride over rows
__global__ void k_bn_stats(const float* __restrict__ x, float* __restrict__ sums, int n) {
    int c = threadIdx.x;
    float s1 = 0.0f, s2 = 0.0f;
    for (int r = blockIdx.x; r < n; r += gridDim.x) {
        float v = x[(size_t)r * 128 + c];
        s1 += v; s2 += v * v;
    }
    atomicAdd(&sums[c], s1);
    atomicAdd(&sums[128 + c], s2);
}

// y_bf16 = bf16(relu(g*(x-mu)*rsqrt(var+eps)+be))   -- feeds next WMMA GEMM directly
__global__ void k_bn_apply_relu_bf(const float* __restrict__ x, const float* __restrict__ sums,
                                   const float* __restrict__ g, const float* __restrict__ be,
                                   unsigned short* __restrict__ y, int n) {
    int idx = blockIdx.x * blockDim.x + threadIdx.x;
    if (idx >= n * 128) return;
    int c = idx & 127;
    float inv = 1.0f / (float)n;
    float mu  = sums[c] * inv;
    float var = sums[128 + c] * inv - mu * mu;
    float v = (x[idx] - mu) * rsqrtf(var + 1e-5f) * g[c] + be[c];
    y[idx] = f2bf(fmaxf(v, 0.0f));
}

// GRU gates: h = (1-z)*tanh(in + r*hn) + z*h ; writes f32 h (in-place) and bf16 mirror
__global__ void k_gru_gates(const float* __restrict__ gi, const float* __restrict__ gh,
                            float* __restrict__ h, unsigned short* __restrict__ h_bf, int n) {
    int idx = blockIdx.x * blockDim.x + threadIdx.x;
    if (idx >= n * 128) return;
    int i = idx >> 7, c = idx & 127;
    const float* gir = gi + (size_t)i * 384;
    const float* ghr = gh + (size_t)i * 384;
    float r  = 1.0f / (1.0f + __expf(-(gir[c]       + ghr[c])));
    float z  = 1.0f / (1.0f + __expf(-(gir[128 + c] + ghr[128 + c])));
    float nc = tanhf(gir[256 + c] + r * ghr[256 + c]);
    float hp = h[idx];
    float hn = (1.0f - z) * nc + z * hp;
    h[idx] = hn;
    h_bf[idx] = f2bf(hn);
}

// out = log_softmax(h[N,128] @ Wout[128,16] + bout)
__global__ void k_out_logsoftmax(const float* __restrict__ h, const float* __restrict__ Wout,
                                 const float* __restrict__ bout, float* __restrict__ out, int n) {
    __shared__ float w[128 * 16];
    for (int i = threadIdx.x; i < 128 * 16; i += blockDim.x) w[i] = Wout[i];
    __syncthreads();
    int i = blockIdx.x * blockDim.x + threadIdx.x;
    if (i >= n) return;
    float logits[16];
    #pragma unroll
    for (int c = 0; c < 16; ++c) logits[c] = bout[c];
    const float* hr = h + (size_t)i * 128;
    for (int k = 0; k < 128; ++k) {
        float hv = hr[k];
        const float* wr = &w[k * 16];
        #pragma unroll
        for (int c = 0; c < 16; ++c) logits[c] += hv * wr[c];
    }
    float mx = logits[0];
    #pragma unroll
    for (int c = 1; c < 16; ++c) mx = fmaxf(mx, logits[c]);
    float s = 0.0f;
    #pragma unroll
    for (int c = 0; c < 16; ++c) s += __expf(logits[c] - mx);
    float lse = mx + __logf(s);
    #pragma unroll
    for (int c = 0; c < 16; ++c) out[(size_t)i * 16 + c] = logits[c] - lse;
}

// ---------------- host orchestration ----------------
static inline void launch_gemm(const unsigned short* A, const unsigned short* B, const float* bias,
                               void* C, int M, int N, int K, int relu, int c_bf16, hipStream_t s) {
    dim3 grid((M + TBM - 1) / TBM, N / TBN);
    gemm_bf16_wmma<<<grid, 256, 0, s>>>(A, B, bias, C, M, N, K, relu, c_bf16);
}

extern "C" void kernel_launch(void* const* d_in, const int* in_sizes, int n_in,
                              void* d_out, int out_size, void* d_ws, size_t ws_size,
                              hipStream_t stream)
{
    constexpr int T = 16, Nn = 20000, E = 640000, F = 64, H = 128;
    (void)in_sizes; (void)n_in; (void)out_size; (void)ws_size;

    const float* x_seq = (const float*)d_in[0];
    const int*   ei    = (const int*)  d_in[1];     // [2,E]: src then dst
    const float* ew    = (const float*)d_in[2];
    const float* W_in  = (const float*)d_in[3];
    const float* b_in  = (const float*)d_in[4];
    const float* W1    = (const float*)d_in[5];
    // bc1/bc2 (d_in[6], d_in[10]) cancel exactly inside BN mean-subtraction
    const float* g1    = (const float*)d_in[7];
    const float* be1   = (const float*)d_in[8];
    const float* W2    = (const float*)d_in[9];
    const float* g2    = (const float*)d_in[11];
    const float* be2   = (const float*)d_in[12];
    const float* Wih0  = (const float*)d_in[13];
    const float* Whh0  = (const float*)d_in[14];
    const float* bih0  = (const float*)d_in[15];
    const float* bhh0  = (const float*)d_in[16];
    const float* Wih1  = (const float*)d_in[17];
    const float* Whh1  = (const float*)d_in[18];
    const float* bih1  = (const float*)d_in[19];
    const float* bhh1  = (const float*)d_in[20];
    const float* W_out = (const float*)d_in[21];
    const float* b_out = (const float*)d_in[22];
    float* out = (float*)d_out;

    const int* e_src = ei;
    const int* e_dst = ei + E;

    // ---- workspace carve-out (256B aligned) ----
    char* ws = (char*)d_ws;
    size_t off = 0;
    auto alloc = [&](size_t bytes) -> void* {
        void* p = ws + off;
        off = (off + bytes + 255) & ~(size_t)255;
        return p;
    };
    float* norm      = (float*)alloc((size_t)E * 4);
    float* selfnorm  = (float*)alloc((size_t)Nn * 4);
    float* deg       = (float*)alloc((size_t)Nn * 4);
    float* dinv      = (float*)alloc((size_t)Nn * 4);
    float* bnsums    = (float*)alloc(256 * 4);
    unsigned short* winb  = (unsigned short*)alloc((size_t)F * H * 2);
    unsigned short* w1b   = (unsigned short*)alloc((size_t)H * H * 2);
    unsigned short* w2b   = (unsigned short*)alloc((size_t)H * H * 2);
    unsigned short* wih0T = (unsigned short*)alloc((size_t)3 * H * H * 2);
    unsigned short* whh0T = (unsigned short*)alloc((size_t)3 * H * H * 2);
    unsigned short* wih1T = (unsigned short*)alloc((size_t)3 * H * H * 2);
    unsigned short* whh1T = (unsigned short*)alloc((size_t)3 * H * H * 2);
    unsigned short* x_bf  = (unsigned short*)alloc((size_t)Nn * F * 2);
    unsigned short* abf   = (unsigned short*)alloc((size_t)Nn * H * 2);   // bf16 activation
    unsigned short* h0bf  = (unsigned short*)alloc((size_t)Nn * H * 2);
    unsigned short* h1bf  = (unsigned short*)alloc((size_t)Nn * H * 2);
    float* bufXW  = (float*)alloc((size_t)Nn * H * 4);
    float* bufAgg = (float*)alloc((size_t)Nn * H * 4);
    float* gi     = (float*)alloc((size_t)Nn * 3 * H * 4);
    float* gh     = (float*)alloc((size_t)Nn * 3 * H * 4);
    float* h0     = (float*)alloc((size_t)Nn * H * 4);
    float* h1     = (float*)alloc((size_t)Nn * H * 4);

    const int TPB = 256;
    auto blocks = [&](int n) { return (n + TPB - 1) / TPB; };

    // ---- weight conversion ----
    k_conv_bf16  <<<blocks(F * H), TPB, 0, stream>>>(W_in, winb, F * H);
    k_conv_bf16  <<<blocks(H * H), TPB, 0, stream>>>(W1, w1b, H * H);
    k_conv_bf16  <<<blocks(H * H), TPB, 0, stream>>>(W2, w2b, H * H);
    k_conv_bf16_T<<<blocks(3 * H * H), TPB, 0, stream>>>(Wih0, wih0T, 3 * H, H);
    k_conv_bf16_T<<<blocks(3 * H * H), TPB, 0, stream>>>(Whh0, whh0T, 3 * H, H);
    k_conv_bf16_T<<<blocks(3 * H * H), TPB, 0, stream>>>(Wih1, wih1T, 3 * H, H);
    k_conv_bf16_T<<<blocks(3 * H * H), TPB, 0, stream>>>(Whh1, whh1T, 3 * H, H);

    // ---- GCN normalization (self-loops folded into deg init) ----
    k_init_f32 <<<blocks(Nn), TPB, 0, stream>>>(deg, 1.0f, Nn);
    k_deg_accum<<<blocks(E), TPB, 0, stream>>>(e_dst, ew, deg, E);
    k_dinv_self<<<blocks(Nn), TPB, 0, stream>>>(deg, dinv, selfnorm, Nn);
    k_edge_norm<<<blocks(E), TPB, 0, stream>>>(e_src, e_dst, ew, dinv, norm, E);

    // ---- GRU hidden init (f32 + bf16 mirrors) ----
    k_init_f32<<<blocks(Nn * H), TPB, 0, stream>>>(h0, 0.0f, Nn * H);
    k_init_f32<<<blocks(Nn * H), TPB, 0, stream>>>(h1, 0.0f, Nn * H);
    k_init_u16<<<blocks(Nn * H), TPB, 0, stream>>>(h0bf, 0, Nn * H);
    k_init_u16<<<blocks(Nn * H), TPB, 0, stream>>>(h1bf, 0, Nn * H);

    for (int t = 0; t < T; ++t) {
        const float* x_t = x_seq + (size_t)t * Nn * F;

        // input features -> bf16, projection + ReLU (bf16 out feeds next GEMM)
        k_conv_bf16<<<blocks(Nn * F), TPB, 0, stream>>>(x_t, x_bf, Nn * F);
        launch_gemm(x_bf, winb, b_in, abf, Nn, H, F, 1, 1, stream);

        // GCN conv 1
        launch_gemm(abf, w1b, nullptr, bufXW, Nn, H, H, 0, 0, stream);
        k_scatter_init <<<blocks(Nn * 32), TPB, 0, stream>>>(bufXW, selfnorm, bufAgg, Nn);
        k_scatter_edges<<<blocks(E * 32), TPB, 0, stream>>>(e_src, e_dst, norm, bufXW, bufAgg, E);
        k_init_f32     <<<1, 256, 0, stream>>>(bnsums, 0.0f, 256);
        k_bn_stats     <<<512, 128, 0, stream>>>(bufAgg, bnsums, Nn);
        k_bn_apply_relu_bf<<<blocks(Nn * H), TPB, 0, stream>>>(bufAgg, bnsums, g1, be1, abf, Nn);

        // GCN conv 2
        launch_gemm(abf, w2b, nullptr, bufXW, Nn, H, H, 0, 0, stream);
        k_scatter_init <<<blocks(Nn * 32), TPB, 0, stream>>>(bufXW, selfnorm, bufAgg, Nn);
        k_scatter_edges<<<blocks(E * 32), TPB, 0, stream>>>(e_src, e_dst, norm, bufXW, bufAgg, E);
        k_init_f32     <<<1, 256, 0, stream>>>(bnsums, 0.0f, 256);
        k_bn_stats     <<<512, 128, 0, stream>>>(bufAgg, bnsums, Nn);
        k_bn_apply_relu_bf<<<blocks(Nn * H), TPB, 0, stream>>>(bufAgg, bnsums, g2, be2, abf, Nn);

        // GRU layer 0 (abf = bf16(s_t))
        launch_gemm(abf,  wih0T, bih0, gi, Nn, 3 * H, H, 0, 0, stream);
        launch_gemm(h0bf, whh0T, bhh0, gh, Nn, 3 * H, H, 0, 0, stream);
        k_gru_gates<<<blocks(Nn * H), TPB, 0, stream>>>(gi, gh, h0, h0bf, Nn);

        // GRU layer 1 (input = h0)
        launch_gemm(h0bf, wih1T, bih1, gi, Nn, 3 * H, H, 0, 0, stream);
        launch_gemm(h1bf, whh1T, bhh1, gh, Nn, 3 * H, H, 0, 0, stream);
        k_gru_gates<<<blocks(Nn * H), TPB, 0, stream>>>(gi, gh, h1, h1bf, Nn);
    }

    // ---- output head + log_softmax ----
    k_out_logsoftmax<<<blocks(Nn), TPB, 0, stream>>>(h1, W_out, b_out, out, Nn);
}
